// MoE_MLP_41918880809116
// MI455X (gfx1250) — matrix operand order repaired and checked by reading, not compiled, not run on hardware
//
#include <hip/hip_runtime.h>
#include <math.h>

typedef __attribute__((ext_vector_type(16))) _Float16 v16h;
typedef __attribute__((ext_vector_type(8)))  _Float16 v8h;
typedef __attribute__((ext_vector_type(8)))  float    v8f;

#define T_TOK 32768
#define D_DIM 256
#define H_DIM 1024
#define E_NUM 8
#define M_TILE 64

// workspace layout (bytes)
#define XH_OFF   0ul                                  // T*D f16      = 16 MB
#define W1T_OFF  16777216ul                           // E*H*D f16    = 4  MB
#define W2T_OFF  (16777216ul + 4194304ul)             // E*D*H f16    = 4  MB
#define WD_OFF   (16777216ul + 8388608ul)             // T*E f32      = 1  MB

__device__ __forceinline__ v16h mkv16(v8h lo, v8h hi) {
  return __builtin_shufflevector(lo, hi, 0,1,2,3,4,5,6,7,8,9,10,11,12,13,14,15);
}

// ---------------- prep: fp32 -> f16 conversions ----------------
__global__ void cvt_x_kernel(const float* __restrict__ x, _Float16* __restrict__ xh, int n) {
  int i = blockIdx.x * 256 + threadIdx.x;
  if (i < n) xh[i] = (_Float16)x[i];
}

// W1 [E][D][H] -> W1t [E][H][D] f16  (K=D contiguous for B fragments of GEMM1)
__global__ void cvt_w1t_kernel(const float* __restrict__ W1, _Float16* __restrict__ W1t, int n) {
  int i = blockIdx.x * 256 + threadIdx.x;
  if (i >= n) return;
  int e = i / (H_DIM * D_DIM);
  int r = i - e * (H_DIM * D_DIM);
  int h = r / D_DIM;
  int d = r - h * D_DIM;
  W1t[i] = (_Float16)W1[(long)e * D_DIM * H_DIM + (long)d * H_DIM + h];
}

// W2 [E][H][D] -> W2t [E][D][H] f16  (K=H contiguous for B fragments of GEMM2)
__global__ void cvt_w2t_kernel(const float* __restrict__ W2, _Float16* __restrict__ W2t, int n) {
  int i = blockIdx.x * 256 + threadIdx.x;
  if (i >= n) return;
  int e = i / (D_DIM * H_DIM);
  int r = i - e * (D_DIM * H_DIM);
  int d = r / H_DIM;
  int h = r - d * H_DIM;
  W2t[i] = (_Float16)W2[(long)e * H_DIM * D_DIM + (long)h * D_DIM + d];
}

// ---------------- gating: logits -> top-2 softmax -> dense [T,E] weights ----------------
__global__ void gate_topk_kernel(const float* __restrict__ x, const float* __restrict__ Wg,
                                 const float* __restrict__ bg, float* __restrict__ wd) {
  __shared__ float sW[D_DIM * E_NUM];
  __shared__ float sB[E_NUM];
  int tid = threadIdx.x;
  for (int i = tid; i < D_DIM * E_NUM; i += 256) sW[i] = Wg[i];
  if (tid < E_NUM) sB[tid] = bg[tid];
  __syncthreads();

  int t = blockIdx.x * 256 + tid;
  float acc[E_NUM];
#pragma unroll
  for (int e = 0; e < E_NUM; ++e) acc[e] = sB[e];

  const float4* xr = (const float4*)(x + (long)t * D_DIM);
#pragma unroll 4
  for (int d4 = 0; d4 < D_DIM / 4; ++d4) {
    float4 v = xr[d4];
    float c[4] = {v.x, v.y, v.z, v.w};
    int d = d4 * 4;
#pragma unroll
    for (int j = 0; j < 4; ++j)
#pragma unroll
      for (int e = 0; e < E_NUM; ++e)
        acc[e] = fmaf(c[j], sW[(d + j) * E_NUM + e], acc[e]);
  }

  float m0 = -INFINITY, m1 = -INFINITY;
  int i0 = 0, i1 = 0;
#pragma unroll
  for (int e = 0; e < E_NUM; ++e) {
    float l = acc[e];
    if (l > m0) { m1 = m0; i1 = i0; m0 = l; i0 = e; }
    else if (l > m1) { m1 = l; i1 = e; }
  }
  float ew = expf(m1 - m0);
  float inv = 1.0f / (1.0f + ew);
  float w0 = inv, w1 = ew * inv;

  float* o = wd + (long)t * E_NUM;
#pragma unroll
  for (int e = 0; e < E_NUM; ++e) o[e] = 0.0f;
  o[i0] = w0;
  o[i1] = w1;
}

// ---------------- main: 64-token tile, dense over experts, fused 2-GEMM + gelu ----------------
// Gate weight is folded into h before GEMM-2:  w*(gelu(xW1+b1)@W2) == (diag(w)*gelu(..))@W2,
// so phase-2 accumulators hold the final weighted sum across experts directly.
// sH is double-buffered -> one barrier per H-chunk instead of two.
__global__ __launch_bounds__(256) void moe_main_kernel(
    const _Float16* __restrict__ xh, const _Float16* __restrict__ W1t,
    const _Float16* __restrict__ W2t, const float* __restrict__ wd,
    const float* __restrict__ b1, const float* __restrict__ b2,
    float* __restrict__ out) {
  __shared__ _Float16 sX[M_TILE * D_DIM];      // 32 KB: x tile, resident all kernel
  __shared__ _Float16 sH[2][M_TILE * 128];     // 2x16 KB: w-scaled gelu(h) chunk (double buffer)
  __shared__ float    sWd[M_TILE * E_NUM];     //  2 KB: gate combine weights

  int tid  = threadIdx.x;
  int wv   = tid >> 5;          // wave 0..7
  int lane = tid & 31;
  int lm   = lane & 15;
  int sel8  = (lane < 16) ? 0 : 8;   // M-half select for C/D rows
  int sel16 = sel8 * 2;              // K-half select for B frags

  long tbase = (long)blockIdx.x * M_TILE;

  // stage x tile (rows contiguous: row length == D) and gate-weight tile
  {
    const uint4* src = (const uint4*)(xh + tbase * D_DIM);
    uint4* dst = (uint4*)sX;
#pragma unroll
    for (int i = 0; i < (M_TILE * D_DIM) / 8 / 256; ++i)
      dst[tid + i * 256] = src[tid + i * 256];
    const float* wsrc = wd + tbase * E_NUM;
#pragma unroll
    for (int i = 0; i < (M_TILE * E_NUM) / 256; ++i)
      sWd[tid + i * 256] = wsrc[tid + i * 256];
  }
  __syncthreads();

  v8f outAcc[4][2];
#pragma unroll
  for (int mt = 0; mt < 4; ++mt) {
    outAcc[mt][0] = {};
    outAcc[mt][1] = {};
  }

  int cc = 0;  // global chunk counter -> sH buffer parity
  for (int e = 0; e < E_NUM; ++e) {
    const _Float16* w1e = W1t + (long)e * H_DIM * D_DIM;
    const _Float16* w2e = W2t + (long)e * D_DIM * H_DIM;

    for (int hc = 0; hc < H_DIM / 128; ++hc, ++cc) {
      _Float16* hbuf = sH[cc & 1];

      // ---- phase 1: hbuf[64 x 128] = w .* gelu(x @ W1[e] + b1), this wave's 16 cols ----
      int hglob = hc * 128 + wv * 16 + lm;                    // this lane's h column
      const _Float16* brow1 = w1e + (long)hglob * D_DIM + sel16;
      v8f p[4];
#pragma unroll
      for (int mt = 0; mt < 4; ++mt) p[mt] = {};
#pragma unroll
      for (int ks = 0; ks < 8; ++ks) {
        int k0 = ks * 32;
        v16h b = *(const v16h*)(brow1 + k0);
        v16h a[4];
#pragma unroll
        for (int mt = 0; mt < 4; ++mt) {
          const _Float16* arow = sX + (mt * 16 + lm) * D_DIM + k0 + sel16;
          a[mt] = mkv16(*(const v8h*)arow, *(const v8h*)(arow + 8));
        }
#pragma unroll
        for (int mt = 0; mt < 4; ++mt)
          p[mt] = __builtin_amdgcn_wmma_f32_16x16x32_f16(false, a[mt], false, b, (short)0, p[mt], false, false);
      }
      float b1v = b1[e * H_DIM + hglob];
#pragma unroll
      for (int mt = 0; mt < 4; ++mt) {
#pragma unroll
        for (int r = 0; r < 8; ++r) {
          int row = mt * 16 + r + sel8;
          float v = p[mt][r] + b1v;
          float g = 0.5f * v * (1.0f + erff(v * 0.70710678118654752f));
          g *= sWd[row * E_NUM + e];
          hbuf[row * 128 + wv * 16 + lm] = (_Float16)g;
        }
      }
      __syncthreads();  // publish h chunk (double buffer -> no trailing barrier needed)

      // ---- phase 2: outAcc += hbuf @ W2[e][hc*128 : hc*128+128, :] ----
      int n0 = wv * 32 + lm;  // wave owns output cols [wv*32, wv*32+32)
#pragma unroll
      for (int ks = 0; ks < 4; ++ks) {
        int kk = ks * 32;
        long kb = (long)(hc * 128 + kk + sel16);
        v16h bA = *(const v16h*)(w2e + (long)n0 * H_DIM + kb);
        v16h bB = *(const v16h*)(w2e + (long)(n0 + 16) * H_DIM + kb);
        v16h a[4];
#pragma unroll
        for (int mt = 0; mt < 4; ++mt) {
          const _Float16* arow = hbuf + (mt * 16 + lm) * 128 + kk + sel16;
          a[mt] = mkv16(*(const v8h*)arow, *(const v8h*)(arow + 8));
        }
#pragma unroll
        for (int mt = 0; mt < 4; ++mt) {
          outAcc[mt][0] = __builtin_amdgcn_wmma_f32_16x16x32_f16(false, a[mt], false, bA, (short)0, outAcc[mt][0], false, false);
          outAcc[mt][1] = __builtin_amdgcn_wmma_f32_16x16x32_f16(false, a[mt], false, bB, (short)0, outAcc[mt][1], false, false);
        }
      }
      // no barrier here: next chunk writes the other sH buffer; the per-chunk
      // barrier above orders chunk c+2 writers against chunk c readers.
    }
  }

  // ---- add bias term sum_e w[t,e]*b2[e,n] and store 64 x 256 fp32 output tile ----
  float b2v0[E_NUM], b2v1[E_NUM];
#pragma unroll
  for (int e = 0; e < E_NUM; ++e) {
    b2v0[e] = b2[e * D_DIM + wv * 32 + lm];
    b2v1[e] = b2[e * D_DIM + wv * 32 + 16 + lm];
  }
#pragma unroll
  for (int mt = 0; mt < 4; ++mt) {
#pragma unroll
    for (int r = 0; r < 8; ++r) {
      int row = mt * 16 + r + sel8;
      float s0 = 0.0f, s1 = 0.0f;
#pragma unroll
      for (int e = 0; e < E_NUM; ++e) {
        float w = sWd[row * E_NUM + e];
        s0 = fmaf(w, b2v0[e], s0);
        s1 = fmaf(w, b2v1[e], s1);
      }
      float* orow = out + (tbase + row) * D_DIM + wv * 32 + lm;
      orow[0]  = outAcc[mt][0][r] + s0;
      orow[16] = outAcc[mt][1][r] + s1;
    }
  }
}

extern "C" void kernel_launch(void* const* d_in, const int* in_sizes, int n_in,
                              void* d_out, int out_size, void* d_ws, size_t ws_size,
                              hipStream_t stream) {
  const float* x  = (const float*)d_in[0];
  const float* Wg = (const float*)d_in[1];
  const float* bg = (const float*)d_in[2];
  const float* W1 = (const float*)d_in[3];
  const float* b1 = (const float*)d_in[4];
  const float* W2 = (const float*)d_in[5];
  const float* b2 = (const float*)d_in[6];

  char* ws = (char*)d_ws;
  _Float16* xh  = (_Float16*)(ws + XH_OFF);
  _Float16* W1t = (_Float16*)(ws + W1T_OFF);
  _Float16* W2t = (_Float16*)(ws + W2T_OFF);
  float*    wd  = (float*)(ws + WD_OFF);

  const int nX = T_TOK * D_DIM;           // 8,388,608
  const int nW = E_NUM * D_DIM * H_DIM;   // 2,097,152

  cvt_x_kernel  <<<(nX + 255) / 256, 256, 0, stream>>>(x,  xh,  nX);
  cvt_w1t_kernel<<<(nW + 255) / 256, 256, 0, stream>>>(W1, W1t, nW);
  cvt_w2t_kernel<<<(nW + 255) / 256, 256, 0, stream>>>(W2, W2t, nW);
  gate_topk_kernel<<<T_TOK / 256, 256, 0, stream>>>(x, Wg, bg, wd);
  moe_main_kernel<<<T_TOK / M_TILE, 256, 0, stream>>>(xh, W1t, W2t, wd, b1, b2, (float*)d_out);
}
